// SelfAttention_17746804867163
// MI455X (gfx1250) — matrix operand order repaired and checked by reading, not compiled, not run on hardware
//
#include <hip/hip_runtime.h>
#include <hip/hip_bf16.h>

#define EMB   1024
#define SEQ   2048
#define BATCH 2
#define NHEAD 16
#define HDIM  64
#define ROWS  (BATCH*SEQ)   /* 4096 */

typedef _Float16 v16h __attribute__((ext_vector_type(16)));
typedef _Float16 v8h  __attribute__((ext_vector_type(8)));
typedef float    v8f  __attribute__((ext_vector_type(8)));

static __device__ __forceinline__ v16h cat8(v8h lo, v8h hi) {
  return __builtin_shufflevector(lo, hi, 0,1,2,3,4,5,6,7,8,9,10,11,12,13,14,15);
}
static __device__ __forceinline__ v16h ldA(const _Float16* p) {
  // A-frag 16x32 f16: lane<16 -> K 0-15, lane>=16 -> K 16-31
  return cat8(*(const v8h*)p, *(const v8h*)(p + 8));
}
static __device__ __forceinline__ v8f wmma16(v16h a, v16h b, v8f c) {
  return __builtin_amdgcn_wmma_f32_16x16x32_f16(false, a, false, b, (short)0, c, false, false);
}

// ---------------------------------------------------------------------------
// fp32 -> f16 conversion (8 elems / thread)
// ---------------------------------------------------------------------------
__global__ __launch_bounds__(256) void cvt_f16(const float* __restrict__ s,
                                               _Float16* __restrict__ d, int n8) {
  int i = blockIdx.x * blockDim.x + threadIdx.x;
  if (i < n8) {
    float4 f0 = ((const float4*)s)[2 * i];
    float4 f1 = ((const float4*)s)[2 * i + 1];
    v8h h = { (_Float16)f0.x, (_Float16)f0.y, (_Float16)f0.z, (_Float16)f0.w,
              (_Float16)f1.x, (_Float16)f1.y, (_Float16)f1.z, (_Float16)f1.w };
    ((v8h*)d)[i] = h;
  }
}

// ---------------------------------------------------------------------------
// QKV projection: y = x @ W^T.  W row-major [out,in] == column-major B.
// 32 rows x 64 cols per wave.  Ping-pong fragment buffers (x2 unrolled
// k-loop): no cur<-next register rotation, loads of one buffer overlap the
// WMMAs of the other.  blockIdx.y = 0/1/2 -> Q/K/V.
// ---------------------------------------------------------------------------
__global__ __launch_bounds__(256)
void qkv_gemm(const _Float16* __restrict__ X,
              const _Float16* __restrict__ Wq, const _Float16* __restrict__ Wk,
              const _Float16* __restrict__ Wv,
              _Float16* __restrict__ Qd, _Float16* __restrict__ Kd,
              _Float16* __restrict__ Vt) {
  const int lane = threadIdx.x & 31;
  const int wave = threadIdx.x >> 5;
  const int w    = blockIdx.x * 8 + wave;        // 0..2047
  const int row0 = (w >> 4) * 32;
  const int col0 = (w & 15) * 64;
  const int mode = blockIdx.y;
  const _Float16* W = (mode == 0) ? Wq : (mode == 1) ? Wk : Wv;

  const int am = lane & 15;
  const int ak = (lane >> 4) * 16;
  const int bn = lane & 15;
  const int bk = (lane >> 4) * 16;

  const _Float16* a0 = X + (size_t)(row0 + am) * EMB + ak;
  const _Float16* a1 = a0 + (size_t)16 * EMB;
  const _Float16* bp[4];
#pragma unroll
  for (int t = 0; t < 4; ++t)
    bp[t] = W + (size_t)(col0 + t * 16 + bn) * EMB + bk;

  v8f acc[2][4];
#pragma unroll
  for (int m = 0; m < 2; ++m)
#pragma unroll
    for (int t = 0; t < 4; ++t) acc[m][t] = (v8f){};

  // ping-pong buffers: A = even k-steps, B = odd k-steps
  v16h aA0 = ldA(a0),       aA1 = ldA(a1);
  v16h bA[4];
#pragma unroll
  for (int t = 0; t < 4; ++t) bA[t] = *(const v16h*)(bp[t]);
  v16h aB0 = ldA(a0 + 32),  aB1 = ldA(a1 + 32);
  v16h bB[4];
#pragma unroll
  for (int t = 0; t < 4; ++t) bB[t] = *(const v16h*)(bp[t] + 32);

  for (int k0 = 0; k0 < EMB - 64; k0 += 64) {
    __builtin_prefetch(a0 + k0 + 512, 0, 3);
    // compute with A(k0); refill A with k0+64 (overlaps B's WMMAs)
#pragma unroll
    for (int t = 0; t < 4; ++t) {
      acc[0][t] = wmma16(aA0, bA[t], acc[0][t]);
      acc[1][t] = wmma16(aA1, bA[t], acc[1][t]);
    }
    aA0 = ldA(a0 + k0 + 64);
    aA1 = ldA(a1 + k0 + 64);
#pragma unroll
    for (int t = 0; t < 4; ++t) bA[t] = *(const v16h*)(bp[t] + k0 + 64);
    // compute with B(k0+32); refill B with k0+96
#pragma unroll
    for (int t = 0; t < 4; ++t) {
      acc[0][t] = wmma16(aB0, bB[t], acc[0][t]);
      acc[1][t] = wmma16(aB1, bB[t], acc[1][t]);
    }
    aB0 = ldA(a0 + k0 + 96);
    aB1 = ldA(a1 + k0 + 96);
#pragma unroll
    for (int t = 0; t < 4; ++t) bB[t] = *(const v16h*)(bp[t] + k0 + 96);
  }
  // epilogue: A holds k=960, B holds k=992
#pragma unroll
  for (int t = 0; t < 4; ++t) {
    acc[0][t] = wmma16(aA0, bA[t], acc[0][t]);
    acc[1][t] = wmma16(aA1, bA[t], acc[1][t]);
  }
#pragma unroll
  for (int t = 0; t < 4; ++t) {
    acc[0][t] = wmma16(aB0, bB[t], acc[0][t]);
    acc[1][t] = wmma16(aB1, bB[t], acc[1][t]);
  }

  // C-layout: vgpr i -> M = i + 8*(lane>=16), N = lane&15
  const int mbase = (lane >> 4) * 8;
  const int nloc  = lane & 15;
#pragma unroll
  for (int m = 0; m < 2; ++m) {
    int r0 = row0 + m * 16 + mbase;       // 8 consecutive rows r0..r0+7
    int bb = r0 >> 11;
    int tt = r0 & (SEQ - 1);
#pragma unroll
    for (int tile = 0; tile < 4; ++tile) {
      int o = col0 + tile * 16 + nloc;
      int h = o >> 6, d = o & 63;
      if (mode == 2) {
        // V transposed [B,H,S,T]: consecutive i -> consecutive t: b128 store
        v8h pack;
#pragma unroll
        for (int i = 0; i < 8; ++i) pack[i] = (_Float16)acc[m][tile][i];
        *(v8h*)(Vt + ((size_t)(bb * NHEAD + h) * HDIM + d) * SEQ + tt) = pack;
      } else {
        _Float16* dst = ((mode == 0) ? Qd : Kd) +
                        ((size_t)(bb * NHEAD + h) * SEQ + tt) * HDIM + d;
        float s = (mode == 0) ? 0.125f : 1.0f;
#pragma unroll
        for (int i = 0; i < 8; ++i)
          dst[(size_t)i * HDIM] = (_Float16)(acc[m][tile][i] * s);
      }
    }
  }
}

// ---------------------------------------------------------------------------
// Flash attention: one 32-query tile per wave, 32 keys per block step.
// Full (unmasked) blocks j < t0, then one diagonal masked block at j == t0.
// K loads first, V loads second (softmax covers V latency); next block's
// K/V lines prefetched during the current block.
// ---------------------------------------------------------------------------
__global__ __launch_bounds__(256)
void attn_fwd(const _Float16* __restrict__ Q, const _Float16* __restrict__ K,
              const _Float16* __restrict__ V, _Float16* __restrict__ AO) {
  __shared__ _Float16 pbuf[8 * 1024];          // 32x32 f16 per wave
  const int lane = threadIdx.x & 31;
  const int wave = threadIdx.x >> 5;
  const int w    = blockIdx.x * 8 + wave;      // 0..2047
  const int bh   = w >> 6;                     // 0..31
  const int t0   = (w & 63) * 32;              // 32-query tile

  const _Float16* Qp = Q + (size_t)bh * SEQ * HDIM;
  const _Float16* Kp = K + (size_t)bh * SEQ * HDIM;
  const _Float16* Vp = V + (size_t)bh * HDIM * SEQ;
  _Float16* pw = pbuf + wave * 1024;

  const int am    = lane & 15;
  const int ak8   = (lane >> 4) * 16;
  const int nloc  = lane & 15;
  const int mbase = (lane >> 4) * 8;
  const int bk16  = (lane >> 4) * 16;
  const float NEG_INF = -__builtin_inff();

  v16h aQ[2][2];
#pragma unroll
  for (int m = 0; m < 2; ++m) {
    const _Float16* qrow = Qp + (size_t)(t0 + m * 16 + am) * HDIM;
    aQ[m][0] = ldA(qrow + ak8);
    aQ[m][1] = ldA(qrow + 32 + ak8);
  }

  float mr[2][8], lr[2][8];
  v8f oc[2][4];
#pragma unroll
  for (int m = 0; m < 2; ++m) {
#pragma unroll
    for (int i = 0; i < 8; ++i) { mr[m][i] = NEG_INF; lr[m][i] = 0.f; }
#pragma unroll
    for (int t = 0; t < 4; ++t) oc[m][t] = (v8f){};
  }

  auto block_step = [&](int j, bool masked) {
    // ---- K fragments (needed first), then V fragments
    const _Float16* kc0 = Kp + (size_t)(j + nloc) * HDIM + bk16;
    const _Float16* kc1 = Kp + (size_t)(j + 16 + nloc) * HDIM + bk16;
    v16h kb00 = *(const v16h*)(kc0);
    v16h kb01 = *(const v16h*)(kc0 + 32);
    v16h kb10 = *(const v16h*)(kc1);
    v16h kb11 = *(const v16h*)(kc1 + 32);
    v16h vb[4];
#pragma unroll
    for (int dt = 0; dt < 4; ++dt)
      vb[dt] = *(const v16h*)(Vp + (size_t)(dt * 16 + nloc) * SEQ + j + bk16);

    // ---- prefetch next key block (K rows + V stripes)
    if (j + 32 <= t0) {
      __builtin_prefetch(kc0 + (size_t)32 * HDIM, 0, 3);
      __builtin_prefetch(kc1 + (size_t)32 * HDIM, 0, 3);
      __builtin_prefetch(Vp + (size_t)nloc * SEQ + j + 32 + bk16, 0, 3);
      __builtin_prefetch(Vp + (size_t)(32 + nloc) * SEQ + j + 32 + bk16, 0, 3);
    }

    // ---- scores: 2 M-tiles x 2 key-tiles, K-dim 64 in two chunks
    v8f sc[2][2];
#pragma unroll
    for (int m = 0; m < 2; ++m) {
      sc[m][0] = (v8f){}; sc[m][1] = (v8f){};
      sc[m][0] = wmma16(aQ[m][0], kb00, sc[m][0]);
      sc[m][0] = wmma16(aQ[m][1], kb01, sc[m][0]);
      sc[m][1] = wmma16(aQ[m][0], kb10, sc[m][1]);
      sc[m][1] = wmma16(aQ[m][1], kb11, sc[m][1]);
    }

    if (masked) {
      int k0v = j + nloc, k1v = j + 16 + nloc;
#pragma unroll
      for (int m = 0; m < 2; ++m)
#pragma unroll
        for (int i = 0; i < 8; ++i) {
          int rq = t0 + m * 16 + mbase + i;
          if (k0v > rq) sc[m][0][i] = NEG_INF;
          if (k1v > rq) sc[m][1][i] = NEG_INF;
        }
    }

    // ---- online softmax (row = vgpr idx + half-wave, 16-lane reductions)
#pragma unroll
    for (int m = 0; m < 2; ++m) {
#pragma unroll
      for (int i = 0; i < 8; ++i) {
        float v0 = sc[m][0][i], v1 = sc[m][1][i];
        float mx = fmaxf(v0, v1);
        mx = fmaxf(mx, __shfl_xor(mx, 1, 32));
        mx = fmaxf(mx, __shfl_xor(mx, 2, 32));
        mx = fmaxf(mx, __shfl_xor(mx, 4, 32));
        mx = fmaxf(mx, __shfl_xor(mx, 8, 32));
        float mnew  = fmaxf(mr[m][i], mx);
        float scale = __expf(mr[m][i] - mnew);
        float p0 = __expf(v0 - mnew);
        float p1 = __expf(v1 - mnew);
        float rs = p0 + p1;
        rs += __shfl_xor(rs, 1, 32);
        rs += __shfl_xor(rs, 2, 32);
        rs += __shfl_xor(rs, 4, 32);
        rs += __shfl_xor(rs, 8, 32);
        lr[m][i] = lr[m][i] * scale + rs;
        mr[m][i] = mnew;
        oc[m][0][i] *= scale; oc[m][1][i] *= scale;
        oc[m][2][i] *= scale; oc[m][3][i] *= scale;
        int mm = m * 16 + mbase + i;
        pw[mm * 32 + nloc]      = (_Float16)p0;   // row-major 32x32 P in LDS
        pw[mm * 32 + 16 + nloc] = (_Float16)p1;
      }
    }

    // ---- P(32x32) . V(32x64)
#pragma unroll
    for (int m = 0; m < 2; ++m) {
      v16h aP = ldA(pw + (m * 16 + am) * 32 + ak8);
#pragma unroll
      for (int dt = 0; dt < 4; ++dt)
        oc[m][dt] = wmma16(aP, vb[dt], oc[m][dt]);
    }
  };

  for (int j = 0; j < t0; j += 32) block_step(j, false);
  block_step(t0, true);

  // ---- normalize and write attention output (f16, [B,T,C])
  const int bb = bh >> 4, h = bh & 15;
#pragma unroll
  for (int m = 0; m < 2; ++m) {
#pragma unroll
    for (int i = 0; i < 8; ++i) {
      float inv = 1.0f / lr[m][i];
      int t = t0 + m * 16 + mbase + i;
      _Float16* dst = AO + (size_t)(bb * SEQ + t) * EMB + h * HDIM + nloc;
#pragma unroll
      for (int tile = 0; tile < 4; ++tile)
        dst[tile * 16] = (_Float16)(oc[m][tile][i] * inv);
    }
  }
}

// ---------------------------------------------------------------------------
// Output projection: y = AO @ Wo^T + bo  (fp32 out), 32x64 per wave,
// ping-pong buffered like qkv_gemm.
// ---------------------------------------------------------------------------
__global__ __launch_bounds__(256)
void out_proj(const _Float16* __restrict__ A, const _Float16* __restrict__ W,
              const float* __restrict__ bo, float* __restrict__ Y) {
  const int lane = threadIdx.x & 31;
  const int wave = threadIdx.x >> 5;
  const int w    = blockIdx.x * 8 + wave;
  const int row0 = (w >> 4) * 32;
  const int col0 = (w & 15) * 64;

  const int am = lane & 15;
  const int ak = (lane >> 4) * 16;
  const int bn = lane & 15;
  const int bk = (lane >> 4) * 16;

  const _Float16* a0 = A + (size_t)(row0 + am) * EMB + ak;
  const _Float16* a1 = a0 + (size_t)16 * EMB;
  const _Float16* bp[4];
#pragma unroll
  for (int t = 0; t < 4; ++t)
    bp[t] = W + (size_t)(col0 + t * 16 + bn) * EMB + bk;

  v8f acc[2][4];
#pragma unroll
  for (int m = 0; m < 2; ++m)
#pragma unroll
    for (int t = 0; t < 4; ++t) acc[m][t] = (v8f){};

  v16h aA0 = ldA(a0),       aA1 = ldA(a1);
  v16h bA[4];
#pragma unroll
  for (int t = 0; t < 4; ++t) bA[t] = *(const v16h*)(bp[t]);
  v16h aB0 = ldA(a0 + 32),  aB1 = ldA(a1 + 32);
  v16h bB[4];
#pragma unroll
  for (int t = 0; t < 4; ++t) bB[t] = *(const v16h*)(bp[t] + 32);

  for (int k0 = 0; k0 < EMB - 64; k0 += 64) {
    __builtin_prefetch(a0 + k0 + 512, 0, 3);
#pragma unroll
    for (int t = 0; t < 4; ++t) {
      acc[0][t] = wmma16(aA0, bA[t], acc[0][t]);
      acc[1][t] = wmma16(aA1, bA[t], acc[1][t]);
    }
    aA0 = ldA(a0 + k0 + 64);
    aA1 = ldA(a1 + k0 + 64);
#pragma unroll
    for (int t = 0; t < 4; ++t) bA[t] = *(const v16h*)(bp[t] + k0 + 64);
#pragma unroll
    for (int t = 0; t < 4; ++t) {
      acc[0][t] = wmma16(aB0, bB[t], acc[0][t]);
      acc[1][t] = wmma16(aB1, bB[t], acc[1][t]);
    }
    aB0 = ldA(a0 + k0 + 96);
    aB1 = ldA(a1 + k0 + 96);
#pragma unroll
    for (int t = 0; t < 4; ++t) bB[t] = *(const v16h*)(bp[t] + k0 + 96);
  }
#pragma unroll
  for (int t = 0; t < 4; ++t) {
    acc[0][t] = wmma16(aA0, bA[t], acc[0][t]);
    acc[1][t] = wmma16(aA1, bA[t], acc[1][t]);
  }
#pragma unroll
  for (int t = 0; t < 4; ++t) {
    acc[0][t] = wmma16(aB0, bB[t], acc[0][t]);
    acc[1][t] = wmma16(aB1, bB[t], acc[1][t]);
  }

  const int mbase = (lane >> 4) * 8;
  const int nloc  = lane & 15;
#pragma unroll
  for (int m = 0; m < 2; ++m) {
#pragma unroll
    for (int tile = 0; tile < 4; ++tile) {
      int o = col0 + tile * 16 + nloc;
      float bias = bo[o];
#pragma unroll
      for (int i = 0; i < 8; ++i) {
        int r = row0 + m * 16 + mbase + i;
        Y[(size_t)r * EMB + o] = acc[m][tile][i] + bias;
      }
    }
  }
}

// ---------------------------------------------------------------------------
extern "C" void kernel_launch(void* const* d_in, const int* in_sizes, int n_in,
                              void* d_out, int out_size, void* d_ws, size_t ws_size,
                              hipStream_t stream) {
  (void)in_sizes; (void)n_in; (void)out_size; (void)ws_size;
  const float* x  = (const float*)d_in[0];
  const float* Wq = (const float*)d_in[1];
  const float* Wk = (const float*)d_in[2];
  const float* Wv = (const float*)d_in[3];
  const float* Wo = (const float*)d_in[4];
  const float* bo = (const float*)d_in[5];
  float* y = (float*)d_out;

  const size_t nX = (size_t)ROWS * EMB;   // 4194304
  const size_t nW = (size_t)EMB * EMB;    // 1048576
  char* ws = (char*)d_ws;
  _Float16* xh  = (_Float16*)ws;          // f16 x            8 MB
  _Float16* wqh = xh  + nX;               // f16 weights      2 MB each
  _Float16* wkh = wqh + nW;
  _Float16* wvh = wkh + nW;
  _Float16* woh = wvh + nW;
  _Float16* qh  = woh + nW;               // Q [B,H,T,S]      8 MB
  _Float16* kh  = qh  + nX;               // K [B,H,T,S]      8 MB
  _Float16* vt  = kh  + nX;               // V [B,H,S,T]      8 MB
  _Float16* ao  = vt  + nX;               // attn out [B,T,C] 8 MB

  cvt_f16<<<dim3((unsigned)(nX / 8 / 256)), 256, 0, stream>>>(x,  xh,  (int)(nX / 8));
  cvt_f16<<<dim3((unsigned)(nW / 8 / 256)), 256, 0, stream>>>(Wq, wqh, (int)(nW / 8));
  cvt_f16<<<dim3((unsigned)(nW / 8 / 256)), 256, 0, stream>>>(Wk, wkh, (int)(nW / 8));
  cvt_f16<<<dim3((unsigned)(nW / 8 / 256)), 256, 0, stream>>>(Wv, wvh, (int)(nW / 8));
  cvt_f16<<<dim3((unsigned)(nW / 8 / 256)), 256, 0, stream>>>(Wo, woh, (int)(nW / 8));

  qkv_gemm<<<dim3(256, 3), 256, 0, stream>>>(xh, wqh, wkh, wvh, qh, kh, vt);
  attn_fwd<<<dim3(256), 256, 0, stream>>>(qh, kh, vt, ao);
  out_proj<<<dim3(256), 256, 0, stream>>>(ao, woh, bo, y);
}
